// MLAttention_22986664968202
// MI455X (gfx1250) — matrix operand order repaired and checked by reading, not compiled, not run on hardware
//
#include <hip/hip_runtime.h>
#include <hip/hip_bf16.h>
#include <math.h>

typedef __attribute__((ext_vector_type(16))) _Float16 v16h;
typedef __attribute__((ext_vector_type(8)))  float    v8f;
typedef __attribute__((ext_vector_type(4)))  unsigned v4u;
typedef __attribute__((ext_vector_type(8)))  unsigned v8u;

static constexpr int S_LEN  = 2048;
static constexpr int NHEADS = 32;
static constexpr int DMODEL = 4096;
static constexpr int QLORA  = 1536;
static constexpr int KVLORA = 512;
static constexpr int ROPE_D = 64;
static constexpr int NOPE_D = 192;
static constexpr int QK_D   = 256;   // NOPE + ROPE
static constexpr int VDIM   = 256;
static constexpr int IDXH   = 4;
static constexpr int IDXD   = 64;
static constexpr int TOPK_K = 512;
static constexpr float NEGINF = -1e30f;
static constexpr float EPSV   = 1e-6f;

union FragU { v16h v; uint4 q[2]; };
union H8    { _Float16 h[8]; uint4 q; };

static __device__ __forceinline__ v8f wmma_f16(v16h a, v16h b, v8f c) {
  return __builtin_amdgcn_wmma_f32_16x16x32_f16(false, a, false, b, (short)0, c,
                                                false, false);
}

static __device__ __forceinline__ v16h load_afrag_h(const _Float16* row, int kbase, int h16) {
  FragU f;
  f.q[0] = *(const uint4*)(row + kbase + h16 * 8);
  f.q[1] = *(const uint4*)(row + kbase + 16 + h16 * 8);
  return f.v;
}
static __device__ __forceinline__ v16h load_bfrag_h(const _Float16* row, int kbase, int h16) {
  FragU f;
  f.q[0] = *(const uint4*)(row + kbase + h16 * 8);
  f.q[1] = *(const uint4*)(row + kbase + 16 + h16 * 8);
  return f.v;
}

// ---------------------------------------------------------------------------
// f32 -> f16 bulk convert
// ---------------------------------------------------------------------------
__global__ void __launch_bounds__(256)
cvt_f16(const float* __restrict__ x, _Float16* __restrict__ y, long n)
{
  long i = ((long)blockIdx.x * 256 + threadIdx.x) * 8;
  if (i + 8 <= n) {
    float4 a = *(const float4*)(x + i);
    float4 b = *(const float4*)(x + i + 4);
    H8 t;
    t.h[0]=(_Float16)a.x; t.h[1]=(_Float16)a.y; t.h[2]=(_Float16)a.z; t.h[3]=(_Float16)a.w;
    t.h[4]=(_Float16)b.x; t.h[5]=(_Float16)b.y; t.h[6]=(_Float16)b.z; t.h[7]=(_Float16)b.w;
    *(uint4*)(y + i) = t.q;
  } else {
    for (long j = i; j < n; ++j) y[j] = (_Float16)x[j];
  }
}

// ---------------------------------------------------------------------------
// Batched WMMA GEMM:  C[M,N] = A[M,K] * op(B),  B always f16.
//   BT==0 : B stored [N,K] (C = A*B^T);  BT==1 : B stored [K,N] (C = A*B)
//   AF16  : A is f16 (async/TDM double-buffered pipeline)
// tdm: 1 = stage B tiles with the Tensor Data Mover (requires full 64-row tiles)
// epi: 0 = f32 store, 1 = attn epilogue (scale+mask bias, f32), 2 = f16 store
// ---------------------------------------------------------------------------
template<int BT, int AF16>
__global__ void __launch_bounds__(256)
gemm_wmma(const void* __restrict__ Av, const _Float16* __restrict__ Bv,
          void* __restrict__ Cv,
          int M, int N, int K, int lda, int ldb, int ldc,
          long sA, long sB, long sC,
          int tdm, int epi, float scale, const unsigned* __restrict__ maskbuf)
{
  constexpr int PIPE = (AF16 && BT == 0) ? 1 : 0;
  constexpr int NBUF = PIPE ? 2 : 1;
  __shared__ _Float16 As[NBUF][64][40];   // 64B data + 16B pad per row
  __shared__ _Float16 Bs[NBUF][64][40];

  const int z = blockIdx.z;
  const _Float16* Ah = (const _Float16*)Av + (size_t)z * sA;
  const float*    Af = (const float*)Av + (size_t)z * sA;
  const _Float16* Bp = Bv + (size_t)z * sB;
  float*    Cf = (float*)Cv + (size_t)z * sC;
  _Float16* Ch = (_Float16*)Cv + (size_t)z * sC;

  const int n0 = blockIdx.x * 64;
  const int m0 = blockIdx.y * 64;
  const int t  = threadIdx.x;
  const int w  = t >> 5, l = t & 31;
  const int wm = (w & 3) * 16;
  const int wn = (w >> 2) * 32;
  const int h16 = l >> 4, l16 = l & 15;

  v8f acc0 = {}; v8f acc1 = {};

  auto stage_a_async = [&](int k0, int buf) {
    int r = t >> 2, c8 = (t & 3) * 8;
    int gm = m0 + r;
    if (gm < M) {
      unsigned lds = (unsigned)(size_t)&As[buf][r][c8];
      const _Float16* gp = Ah + (size_t)gm * lda + k0 + c8;
      asm volatile("global_load_async_to_lds_b128 %0, %1, off"
                   :: "v"(lds), "v"(gp) : "memory");
    }
  };
  auto stage_b_async = [&](int k0, int buf) {
    int r = t >> 2, c8 = (t & 3) * 8;
    int gn = n0 + r;
    if (gn < N) {
      unsigned lds = (unsigned)(size_t)&Bs[buf][r][c8];
      const _Float16* gp = Bp + (size_t)gn * ldb + k0 + c8;
      asm volatile("global_load_async_to_lds_b128 %0, %1, off"
                   :: "v"(lds), "v"(gp) : "memory");
    }
  };
  // TDM: one DMA moves the whole 64x32 f16 B tile, padding 4 DWORDs after
  // every 16 DWORDs to reproduce the 80B LDS row pitch. Wave 0 issues it
  // (TDM ignores EXEC); completion tracked with TENSORcnt.
  auto stage_b_tdm = [&](int k0, int buf) {
    if (w != 0) return;
    unsigned lds = (unsigned)(size_t)&Bs[buf][0][0];
    unsigned long long ga =
        (unsigned long long)(size_t)(Bp + (size_t)n0 * ldb + k0);
    v4u g0; v8u g1; v4u gz = {0u, 0u, 0u, 0u};
    g0[0] = 1u;                                        // count=1, user mode
    g0[1] = lds;                                       // lds_addr (bytes)
    g0[2] = (unsigned)ga;                              // global_addr[31:0]
    g0[3] = (unsigned)((ga >> 32) & 0x01FFFFFFu) | (2u << 30); // addr[56:32]|type=2
    g1[0] = (1u << 16)      // data_size = 2 bytes
          | (1u << 20)      // pad_enable
          | (3u << 22)      // pad_interval: every 16 DWORDs (64B)
          | (3u << 25);     // pad_amount: 4 DWORDs (16B)
    g1[1] = 0u;                        // tensor_dim0 = 1<<30 (low bits 0)
    g1[2] = 0x4000u;                   // tensor_dim0 hi | tensor_dim1 lo
    g1[3] = 0x4000u | (32u << 16);     // tensor_dim1 hi | tile_dim0 = 32
    g1[4] = 64u;                       // tile_dim1 = 64, tile_dim2 = 0
    g1[5] = (unsigned)ldb;             // tensor_dim0_stride (elements)
    g1[6] = 0u;
    g1[7] = 0u;
    asm volatile("tensor_load_to_lds %0, %1, %2, %3"
                 :: "s"(g0), "s"(g1), "s"(gz), "s"(gz) : "memory");
  };

  // manual stage (PV: A is f32, B needs transpose)
  auto stage_manual = [&](int k0) {
    int r = t >> 2, c8 = (t & 3) * 8;
    int gm = m0 + r;
    if (gm < M) {
      const float* p = Af + (size_t)gm * lda + k0 + c8;
      float4 x = *(const float4*)p;
      float4 y = *(const float4*)(p + 4);
      H8 tmp;
      tmp.h[0]=(_Float16)x.x; tmp.h[1]=(_Float16)x.y; tmp.h[2]=(_Float16)x.z; tmp.h[3]=(_Float16)x.w;
      tmp.h[4]=(_Float16)y.x; tmp.h[5]=(_Float16)y.y; tmp.h[6]=(_Float16)y.z; tmp.h[7]=(_Float16)y.w;
      *(uint4*)&As[0][r][c8] = tmp.q;
    }
    int kk = t >> 3, cb = (t & 7) * 8;
    int gk = k0 + kk;
    for (int e = 0; e < 8; ++e) {
      int gn = n0 + cb + e;
      _Float16 v = (gk < K && gn < N) ? Bp[(size_t)gk * ldb + gn] : (_Float16)0.f;
      Bs[0][cb + e][kk] = v;
    }
  };

  auto consume = [&](int buf) {
    FragU fa;
    int mr = wm + l16;
    fa.q[0] = *(const uint4*)&As[buf][mr][h16 * 8];
    fa.q[1] = *(const uint4*)&As[buf][mr][16 + h16 * 8];
    FragU fb0, fb1;
    int nr = wn + l16;
    fb0.q[0] = *(const uint4*)&Bs[buf][nr][h16 * 8];
    fb0.q[1] = *(const uint4*)&Bs[buf][nr][16 + h16 * 8];
    fb1.q[0] = *(const uint4*)&Bs[buf][nr + 16][h16 * 8];
    fb1.q[1] = *(const uint4*)&Bs[buf][nr + 16][16 + h16 * 8];
    acc0 = wmma_f16(fa.v, fb0.v, acc0);
    acc1 = wmma_f16(fa.v, fb1.v, acc1);
  };

  const int nk = K >> 5;   // K is always a multiple of 32 here
  if (PIPE) {
    stage_a_async(0, 0);
    if (tdm) stage_b_tdm(0, 0); else stage_b_async(0, 0);
    for (int kt = 0; kt < nk; ++kt) {
      if (kt + 1 < nk) {
        int k0n = (kt + 1) << 5, bn = (kt + 1) & 1;
        stage_a_async(k0n, bn);
        if (tdm) {
          stage_b_tdm(k0n, bn);
          asm volatile("s_wait_asynccnt 0x1" ::: "memory");
          if (w == 0) __builtin_amdgcn_s_wait_tensorcnt((short)1);
        } else {
          stage_b_async(k0n, bn);
          asm volatile("s_wait_asynccnt 0x2" ::: "memory");
        }
      } else {
        asm volatile("s_wait_asynccnt 0x0" ::: "memory");
        if (tdm && w == 0) __builtin_amdgcn_s_wait_tensorcnt((short)0);
      }
      __syncthreads();
      consume(kt & 1);
      __syncthreads();
    }
  } else {
    for (int kt = 0; kt < nk; ++kt) {
      stage_manual(kt << 5);
      __syncthreads();
      consume(0);
      __syncthreads();
    }
  }

  for (int jt = 0; jt < 2; ++jt) {
    v8f acc = jt ? acc1 : acc0;
    int n = n0 + wn + jt * 16 + l16;
    for (int j = 0; j < 8; ++j) {
      int m = m0 + wm + h16 * 8 + j;
      if (m < M && n < N) {
        float v = acc[j];
        if (epi == 1) {
          unsigned mw = maskbuf[(size_t)m * 64 + (n >> 5)];
          bool ok = (((mw >> (n & 31)) & 1u) != 0u) && (n <= m);
          Cf[(size_t)m * ldc + n] = v * scale + (ok ? 0.f : NEGINF);
        } else if (epi == 2) {
          Ch[(size_t)m * ldc + n] = (_Float16)v;
        } else {
          Cf[(size_t)m * ldc + n] = v;
        }
      }
    }
  }
}

// ---------------------------------------------------------------------------
// RMSNorm over rows (f32 in, f16 out, strided)
// ---------------------------------------------------------------------------
__global__ void __launch_bounds__(256)
rmsnorm_rows(const float* __restrict__ x, int ldx, const float* __restrict__ w,
             _Float16* __restrict__ y, int ldy, int L)
{
  __shared__ float red[8];
  int row = blockIdx.x, t = threadIdx.x;
  const float* xr = x + (size_t)row * ldx;
  _Float16* yr = y + (size_t)row * ldy;
  float ss = 0.f;
  for (int c = t; c < L; c += 256) { float v = xr[c]; ss += v * v; }
  for (int o = 16; o; o >>= 1) ss += __shfl_xor(ss, o, 32);
  if ((t & 31) == 0) red[t >> 5] = ss;
  __syncthreads();
  float tot = 0.f;
  for (int i = 0; i < 8; ++i) tot += red[i];
  float rs = rsqrtf(tot / (float)L + EPSV);
  for (int c = t; c < L; c += 256) yr[c] = (_Float16)(w[c] * (xr[c] * rs));
}

// ---------------------------------------------------------------------------
__global__ void __launch_bounds__(256)
rope_pack_q(const _Float16* __restrict__ q, const float* __restrict__ cosb,
            const float* __restrict__ sinb, _Float16* __restrict__ query)
{
  int s = blockIdx.x, h = blockIdx.y, c = threadIdx.x;
  const _Float16* qr = q + (size_t)s * (NHEADS * QK_D) + h * QK_D;
  float v;
  if (c < NOPE_D) {
    v = (float)qr[c];
  } else {
    int d = c - NOPE_D;
    float x = (float)qr[c];
    float rot = (d < ROPE_D / 2) ? -(float)qr[NOPE_D + d + ROPE_D / 2]
                                 :  (float)qr[NOPE_D + d - ROPE_D / 2];
    v = x * cosb[(size_t)s * ROPE_D + d] + rot * sinb[(size_t)s * ROPE_D + d];
  }
  query[((size_t)h * S_LEN + s) * QK_D + c] = (_Float16)v;
}

__global__ void __launch_bounds__(64)
rope_kpe(const float* __restrict__ ckv, const float* __restrict__ cosb,
         const float* __restrict__ sinb, float* __restrict__ kpe)
{
  int s = blockIdx.x, d = threadIdx.x;
  const float* pr = ckv + (size_t)s * (KVLORA + ROPE_D) + KVLORA;
  float x = pr[d];
  float rot = (d < ROPE_D / 2) ? -pr[d + ROPE_D / 2] : pr[d - ROPE_D / 2];
  kpe[(size_t)s * ROPE_D + d] = x * cosb[(size_t)s * ROPE_D + d]
                              + rot * sinb[(size_t)s * ROPE_D + d];
}

__global__ void __launch_bounds__(256)
pack_kv(const _Float16* __restrict__ kv, const float* __restrict__ kpe,
        _Float16* __restrict__ key, _Float16* __restrict__ value)
{
  int s = blockIdx.x, h = blockIdx.y, c = threadIdx.x;
  const _Float16* kr = kv + (size_t)s * (NHEADS * (NOPE_D + VDIM)) + h * (NOPE_D + VDIM);
  _Float16 kval = (c < NOPE_D) ? kr[c]
                               : (_Float16)kpe[(size_t)s * ROPE_D + (c - NOPE_D)];
  key[((size_t)h * S_LEN + s) * QK_D + c] = kval;
  value[((size_t)h * S_LEN + s) * VDIM + c] = kr[NOPE_D + c];
}

// ---------------------------------------------------------------------------
__global__ void __launch_bounds__(256)
idx_scores(const _Float16* __restrict__ iq, const _Float16* __restrict__ ik,
           const float* __restrict__ gates, unsigned* __restrict__ ukeys)
{
  __shared__ float gs[64 * IDXH];
  int i0 = blockIdx.y * 64, j0 = blockIdx.x * 64;
  int t = threadIdx.x, w = t >> 5, l = t & 31;
  int h16 = l >> 4, l16 = l & 15;
  gs[t] = gates[(size_t)(i0 + (t >> 2)) * IDXH + (t & 3)];
  __syncthreads();

  int wm = (w & 3) * 16, wn = (w >> 2) * 32;
  const _Float16* arow = iq + (size_t)(i0 + wm + l16) * (IDXH * IDXD);
  v16h af[IDXH][2];
  for (int h = 0; h < IDXH; ++h)
    for (int ks = 0; ks < 2; ++ks)
      af[h][ks] = load_afrag_h(arow, h * IDXD + ks * 32, h16);

  for (int nt = 0; nt < 2; ++nt) {
    int jb = j0 + wn + nt * 16;
    const _Float16* brow = ik + (size_t)(jb + l16) * IDXD;
    v16h bf0 = load_bfrag_h(brow, 0, h16);
    v16h bf1 = load_bfrag_h(brow, 32, h16);
    float comb[8] = {0, 0, 0, 0, 0, 0, 0, 0};
    for (int h = 0; h < IDXH; ++h) {
      v8f acc = {};
      acc = wmma_f16(af[h][0], bf0, acc);
      acc = wmma_f16(af[h][1], bf1, acc);
      for (int j = 0; j < 8; ++j) {
        float g = gs[(wm + h16 * 8 + j) * IDXH + h];
        comb[j] += g * fmaxf(acc[j], 0.f);
      }
    }
    for (int j = 0; j < 8; ++j) {
      int gi = i0 + wm + h16 * 8 + j;
      int gj = jb + l16;
      float v = (gj <= gi) ? comb[j] : NEGINF;
      unsigned b = __float_as_uint(v);
      unsigned u = (b & 0x80000000u) ? ~b : (b | 0x80000000u);
      ukeys[(size_t)gi * S_LEN + gj] = u;
    }
  }
}

// ---------------------------------------------------------------------------
__global__ void __launch_bounds__(256)
topk_mask(const unsigned* __restrict__ ukeys, unsigned* __restrict__ mask)
{
  __shared__ unsigned u[S_LEN];
  __shared__ int red[8];
  int gi = blockIdx.x, t = threadIdx.x;
  for (int c = t; c < S_LEN; c += 256) u[c] = ukeys[(size_t)gi * S_LEN + c];
  __syncthreads();

  int keff = (gi + 1 < TOPK_K) ? (gi + 1) : TOPK_K;
  unsigned thr = 0;
  for (int b = 31; b >= 0; --b) {
    unsigned cand = thr | (1u << b);
    int cnt = 0;
    for (int c = t; c < S_LEN; c += 256) cnt += (u[c] >= cand) ? 1 : 0;
    for (int o = 16; o; o >>= 1) cnt += __shfl_xor(cnt, o, 32);
    if ((t & 31) == 0) red[t >> 5] = cnt;
    __syncthreads();
    int tot = 0;
    for (int i = 0; i < 8; ++i) tot += red[i];
    if (tot >= keff) thr = cand;
    __syncthreads();
  }
  if (t < 64) {
    unsigned bits = 0;
    for (int b = 0; b < 32; ++b) {
      int c = t * 32 + b;
      if ((u[c] >= thr) && (c <= gi)) bits |= (1u << b);
    }
    mask[(size_t)gi * 64 + t] = bits;
  }
}

// ---------------------------------------------------------------------------
__global__ void __launch_bounds__(256)
softmax_rows(float* __restrict__ p)
{
  __shared__ float red[8];
  float* pr = p + (size_t)blockIdx.x * S_LEN;
  int t = threadIdx.x;
  float x[8];
  float mx = -3.4e38f;
  for (int i = 0; i < 8; ++i) { x[i] = pr[t + i * 256]; mx = fmaxf(mx, x[i]); }
  for (int o = 16; o; o >>= 1) mx = fmaxf(mx, __shfl_xor(mx, o, 32));
  if ((t & 31) == 0) red[t >> 5] = mx;
  __syncthreads();
  float m = red[0];
  for (int i = 1; i < 8; ++i) m = fmaxf(m, red[i]);
  __syncthreads();
  float s = 0.f;
  for (int i = 0; i < 8; ++i) { x[i] = __expf(x[i] - m); s += x[i]; }
  for (int o = 16; o; o >>= 1) s += __shfl_xor(s, o, 32);
  if ((t & 31) == 0) red[t >> 5] = s;
  __syncthreads();
  float tot = 0.f;
  for (int i = 0; i < 8; ++i) tot += red[i];
  float inv = 1.f / tot;
  for (int i = 0; i < 8; ++i) pr[t + i * 256] = x[i] * inv;
}

// ---------------------------------------------------------------------------
extern "C" void kernel_launch(void* const* d_in, const int* in_sizes, int n_in,
                              void* d_out, int out_size, void* d_ws, size_t ws_size,
                              hipStream_t stream)
{
  (void)in_sizes; (void)n_in; (void)out_size; (void)ws_size;

  const float* hidden     = (const float*)d_in[0];
  const float* cosb       = (const float*)d_in[1];
  const float* sinb       = (const float*)d_in[2];
  const float* q_a_w      = (const float*)d_in[3];
  const float* q_a_ln_w   = (const float*)d_in[4];
  const float* q_b_w      = (const float*)d_in[5];
  const float* kv_a_w     = (const float*)d_in[6];
  const float* kv_a_ln_w  = (const float*)d_in[7];
  const float* kv_b_w     = (const float*)d_in[8];
  const float* o_w        = (const float*)d_in[9];
  const float* idx_q_w    = (const float*)d_in[10];
  const float* idx_k_w    = (const float*)d_in[11];
  const float* idx_gate_w = (const float*)d_in[12];

  float* out   = (float*)d_out;                 // [S, DMODEL]
  float* probs = out + (size_t)S_LEN * DMODEL;  // [H, S, S] fp32 (mandatory output)

  char* wsb = (char*)d_ws;
  size_t off = 0;
  auto alloc = [&](size_t bytes) -> void* {
    void* p = wsb + off; off = (off + bytes + 255) & ~(size_t)255; return p;
  };

  float*     qa      = (float*)alloc(4ull * S_LEN * QLORA);
  _Float16*  qa16    = (_Float16*)alloc(2ull * S_LEN * QLORA);
  float*     ckv     = (float*)alloc(4ull * S_LEN * (KVLORA + ROPE_D));
  _Float16*  kcomp16 = (_Float16*)alloc(2ull * S_LEN * KVLORA);
  float*     kpe     = (float*)alloc(4ull * S_LEN * ROPE_D);
  _Float16*  q16     = (_Float16*)alloc(2ull * S_LEN * NHEADS * QK_D); // reused as attnP16
  _Float16*  kv16    = (_Float16*)alloc(2ull * S_LEN * NHEADS * (NOPE_D + VDIM));
  _Float16*  query16 = (_Float16*)alloc(2ull * NHEADS * S_LEN * QK_D);
  _Float16*  key16   = (_Float16*)alloc(2ull * NHEADS * S_LEN * QK_D);
  _Float16*  value16 = (_Float16*)alloc(2ull * NHEADS * S_LEN * VDIM);
  _Float16*  iq16    = (_Float16*)alloc(2ull * S_LEN * IDXH * IDXD);
  _Float16*  ik16    = (_Float16*)alloc(2ull * S_LEN * IDXD);
  float*     gates   = (float*)alloc(4ull * S_LEN * IDXH);
  unsigned*  mask    = (unsigned*)alloc(4ull * S_LEN * 64);
  unsigned*  idxu    = (unsigned*)alloc(4ull * S_LEN * S_LEN);
  _Float16*  hidden16 = (_Float16*)alloc(2ull * S_LEN * DMODEL);
  _Float16*  qaw16   = (_Float16*)alloc(2ull * QLORA * DMODEL);
  _Float16*  qbw16   = (_Float16*)alloc(2ull * NHEADS * QK_D * QLORA);
  _Float16*  kvaw16  = (_Float16*)alloc(2ull * (KVLORA + ROPE_D) * DMODEL);
  _Float16*  kvbw16  = (_Float16*)alloc(2ull * NHEADS * (NOPE_D + VDIM) * KVLORA);
  _Float16*  ow16    = (_Float16*)alloc(2ull * DMODEL * NHEADS * VDIM);
  _Float16*  iqw16   = (_Float16*)alloc(2ull * IDXH * IDXD * QLORA);
  _Float16*  ikw16   = (_Float16*)alloc(2ull * IDXD * DMODEL);
  _Float16*  igw16   = (_Float16*)alloc(2ull * IDXH * DMODEL);

  dim3 blk(256);
  auto grd = [](int M, int N, int Z) { return dim3((N + 63) / 64, (M + 63) / 64, Z); };
  auto cvt = [&](const float* x, _Float16* y, long n) {
    cvt_f16<<<(unsigned)((n / 8 + 255) / 256), blk, 0, stream>>>(x, y, n);
  };

  // 0) f32 -> f16 operand conversion (one pass each)
  cvt(hidden, hidden16, (long)S_LEN * DMODEL);
  cvt(q_a_w, qaw16, (long)QLORA * DMODEL);
  cvt(q_b_w, qbw16, (long)NHEADS * QK_D * QLORA);
  cvt(kv_a_w, kvaw16, (long)(KVLORA + ROPE_D) * DMODEL);
  cvt(kv_b_w, kvbw16, (long)NHEADS * (NOPE_D + VDIM) * KVLORA);
  cvt(o_w, ow16, (long)DMODEL * NHEADS * VDIM);
  cvt(idx_q_w, iqw16, (long)IDXH * IDXD * QLORA);
  cvt(idx_k_w, ikw16, (long)IDXD * DMODEL);
  cvt(idx_gate_w, igw16, (long)IDXH * DMODEL);

  // 1) q_a = hidden @ q_a_w^T
  gemm_wmma<0, 1><<<grd(S_LEN, QLORA, 1), blk, 0, stream>>>(
      hidden16, qaw16, qa, S_LEN, QLORA, DMODEL, DMODEL, DMODEL, QLORA,
      0, 0, 0, 1, 0, 1.f, nullptr);
  // 2) q_resid = rmsnorm(q_a) -> f16
  rmsnorm_rows<<<S_LEN, blk, 0, stream>>>(qa, QLORA, q_a_ln_w, qa16, QLORA, QLORA);
  // 3) q = q_resid @ q_b_w^T -> f16
  gemm_wmma<0, 1><<<grd(S_LEN, NHEADS * QK_D, 1), blk, 0, stream>>>(
      qa16, qbw16, q16, S_LEN, NHEADS * QK_D, QLORA, QLORA, QLORA, NHEADS * QK_D,
      0, 0, 0, 1, 2, 1.f, nullptr);
  // 4) ckv = hidden @ kv_a_w^T (f32)
  gemm_wmma<0, 1><<<grd(S_LEN, KVLORA + ROPE_D, 1), blk, 0, stream>>>(
      hidden16, kvaw16, ckv, S_LEN, KVLORA + ROPE_D, DMODEL, DMODEL, DMODEL,
      KVLORA + ROPE_D, 0, 0, 0, 1, 0, 1.f, nullptr);
  // 5) k_comp = rmsnorm(ckv[:, :512]) -> f16
  rmsnorm_rows<<<S_LEN, blk, 0, stream>>>(ckv, KVLORA + ROPE_D, kv_a_ln_w,
                                          kcomp16, KVLORA, KVLORA);
  // 6) k_pe rope (f32)
  rope_kpe<<<S_LEN, 64, 0, stream>>>(ckv, cosb, sinb, kpe);
  // 7) kv = k_comp @ kv_b_w^T -> f16
  gemm_wmma<0, 1><<<grd(S_LEN, NHEADS * (NOPE_D + VDIM), 1), blk, 0, stream>>>(
      kcomp16, kvbw16, kv16, S_LEN, NHEADS * (NOPE_D + VDIM), KVLORA,
      KVLORA, KVLORA, NHEADS * (NOPE_D + VDIM), 0, 0, 0, 1, 2, 1.f, nullptr);
  // 8) pack query (RoPE on pe tail)
  rope_pack_q<<<dim3(S_LEN, NHEADS), blk, 0, stream>>>(q16, cosb, sinb, query16);
  // 9) pack key/value
  pack_kv<<<dim3(S_LEN, NHEADS), blk, 0, stream>>>(kv16, kpe, key16, value16);
  // 10) indexer projections
  gemm_wmma<0, 1><<<grd(S_LEN, IDXH * IDXD, 1), blk, 0, stream>>>(
      qa16, iqw16, iq16, S_LEN, IDXH * IDXD, QLORA, QLORA, QLORA, IDXH * IDXD,
      0, 0, 0, 1, 2, 1.f, nullptr);
  gemm_wmma<0, 1><<<grd(S_LEN, IDXD, 1), blk, 0, stream>>>(
      hidden16, ikw16, ik16, S_LEN, IDXD, DMODEL, DMODEL, DMODEL, IDXD,
      0, 0, 0, 1, 2, 1.f, nullptr);
  gemm_wmma<0, 1><<<grd(S_LEN, IDXH, 1), blk, 0, stream>>>(
      hidden16, igw16, gates, S_LEN, IDXH, DMODEL, DMODEL, DMODEL, IDXH,
      0, 0, 0, 0 /* partial tile: async path */, 0, 1.f, nullptr);
  // 11) indexer scores + exact top-k mask
  idx_scores<<<dim3(S_LEN / 64, S_LEN / 64), blk, 0, stream>>>(iq16, ik16, gates, idxu);
  topk_mask<<<S_LEN, blk, 0, stream>>>(idxu, mask);
  // 12) attention scores (batched over heads) + fused scale/bias epilogue
  gemm_wmma<0, 1><<<grd(S_LEN, S_LEN, NHEADS), blk, 0, stream>>>(
      query16, key16, probs, S_LEN, S_LEN, QK_D, QK_D, QK_D, S_LEN,
      (long)S_LEN * QK_D, (long)S_LEN * QK_D, (long)S_LEN * S_LEN,
      1, 1, 0.0625f /* QK^-0.5 */, mask);
  // 13) softmax in place (probs is the fp32 output)
  softmax_rows<<<NHEADS * S_LEN, blk, 0, stream>>>(probs);
  // 14) attn = probs @ value, head-strided into packed f16 [S, H*256] (reuses q16)
  gemm_wmma<1, 0><<<grd(S_LEN, VDIM, NHEADS), blk, 0, stream>>>(
      probs, value16, q16, S_LEN, VDIM, S_LEN, S_LEN, VDIM, NHEADS * VDIM,
      (long)S_LEN * S_LEN, (long)S_LEN * VDIM, (long)VDIM,
      0, 2, 1.f, nullptr);
  // 15) out = attn_packed @ o_w^T  (f32 out)
  gemm_wmma<0, 1><<<grd(S_LEN, DMODEL, 1), blk, 0, stream>>>(
      q16, ow16, out, S_LEN, DMODEL, NHEADS * VDIM, NHEADS * VDIM, NHEADS * VDIM,
      DMODEL, 0, 0, 0, 1, 0, 1.f, nullptr);
}
